// ProductAttention_68745246540423
// MI455X (gfx1250) — matrix operand order repaired and checked by reading, not compiled, not run on hardware
//
#include <hip/hip_runtime.h>
#include <hip/hip_bf16.h>
#include <math.h>
#include <stdint.h>

// ---------------- problem constants ----------------
#define CC    384          // channels
#define HH    192          // height
#define WW_   192          // width
#define NB    2            // B*T
#define NHEAD 12
#define HD    32           // head dim
#define WSZ   8            // window side
#define NWS   24           // windows per side (192/8)
#define NWIN  (NB*NWS*NWS) // 1152
#define TOKENS (NB*HH*WW_) // 73728
#define HW    (HH*WW_)     // 36864

typedef __attribute__((ext_vector_type(16))) _Float16 v16h;
typedef __attribute__((ext_vector_type(8)))  _Float16 v8h;
typedef __attribute__((ext_vector_type(8)))  float    v8f;

__device__ __forceinline__ v8f wmma_f16(v16h a, v16h b, v8f c) {
  return __builtin_amdgcn_wmma_f32_16x16x32_f16(
      /*neg_a=*/false, a, /*neg_b=*/false, b,
      /*c_mod=*/(short)0, c, /*reuse_a=*/false, /*reuse_b=*/false);
}

// Async global -> LDS copy of 16B (per lane), CDNA5 GLOBAL_LOAD_ASYNC_TO_LDS_B128.
// The immediate offset is added to both the LDS and the global address.
__device__ __forceinline__ void async_b128(unsigned ldsAddr, unsigned long long gAddr) {
  asm volatile("global_load_async_to_lds_b128 %0, %1, off"
               :: "v"(ldsAddr), "v"(gAddr) : "memory");
}
__device__ __forceinline__ void async_b128_off16(unsigned ldsAddr, unsigned long long gAddr) {
  asm volatile("global_load_async_to_lds_b128 %0, %1, off offset:16"
               :: "v"(ldsAddr), "v"(gAddr) : "memory");
}
__device__ __forceinline__ void wait_async0() {
  asm volatile("s_wait_asynccnt 0x0" ::: "memory");
}

// ---------------- kernel 1: f32 -> f16 weight convert ----------------
__global__ void cvt_f16_kernel(const float* __restrict__ s, _Float16* __restrict__ d, int n) {
  int i = blockIdx.x * 256 + threadIdx.x;
  if (i < n) d[i] = (_Float16)s[i];
}

// ---------------- kernel 2: fused depthwise 3x3 for q,k,v ----------------
// x: NCHW f32 -> dwq/dwk/dwv: [token][C] f16 (token = n*HW + h*W + w)
__global__ __launch_bounds__(256)
void dwconv_kernel(const float* __restrict__ x,
                   const float* __restrict__ qdw, const float* __restrict__ qdb,
                   const float* __restrict__ kdw, const float* __restrict__ kdb,
                   const float* __restrict__ vdw, const float* __restrict__ vdb,
                   _Float16* __restrict__ dwq, _Float16* __restrict__ dwk,
                   _Float16* __restrict__ dwv) {
  int tid = blockIdx.x * 256 + threadIdx.x;
  if (tid >= NB * CC * HW) return;
  int w = tid % WW_;
  int h = (tid / WW_) % HH;
  int c = (tid / HW) % CC;
  int n = tid / (HW * CC);
  const float* xp = x + ((size_t)n * CC + c) * HW;
  float aq = qdb[c], ak = kdb[c], av = vdb[c];
#pragma unroll
  for (int ky = 0; ky < 3; ky++) {
    int hy = h + ky - 1;
    if (hy < 0 || hy >= HH) continue;
#pragma unroll
    for (int kx = 0; kx < 3; kx++) {
      int wx = w + kx - 1;
      if (wx < 0 || wx >= WW_) continue;
      float xv = xp[hy * WW_ + wx];
      int wi = c * 9 + ky * 3 + kx;
      aq += xv * qdw[wi];
      ak += xv * kdw[wi];
      av += xv * vdw[wi];
    }
  }
  size_t t = (size_t)n * HW + (size_t)h * WW_ + w;
  dwq[t * CC + c] = (_Float16)aq;
  dwk[t * CC + c] = (_Float16)ak;
  dwv[t * CC + c] = (_Float16)av;
}

// ---------------- kernel 3: WMMA GEMM with async-LDS B staging ----------------
// out[t,o] = (sum_c A[t,c]*Bw[o,c] + bias[o]) * scale
// mode 0: q  -> windowed [win][head][tok64][hd] f16 (scaled)
// mode 1: k  -> windowed [win][head][tok64][hd] f16
// mode 2: v  -> windowed TRANSPOSED [win][head][hd][tok64] f16
// mode 3: proj -> NCHW f32 to d_out
#define BCOL_STRIDE 80  // 64 data bytes (32 halves) + 16B pad: 16B-aligned, conflict-free
__global__ __launch_bounds__(128)
void gemm_wmma_kernel(const _Float16* __restrict__ A,   // [TOKENS][CC]
                      const _Float16* __restrict__ Bw,  // [CC][CC] = w[o][c] f16
                      const float* __restrict__ bias,   // [CC]
                      void* __restrict__ out, int mode, float scale) {
  __shared__ __align__(16) unsigned char ldsB[2][64 * BCOL_STRIDE];  // 2 x 5KB

  const int tid  = threadIdx.x;
  const int lane = tid & 31;
  const int wave = tid >> 5;
  const int mL = lane & 15;
  const int hl = lane >> 4;
  const int rowBlock = blockIdx.x * 64 + wave * 16;  // 16 rows per wave
  const int colBlock = blockIdx.y * 64;
  const int row = rowBlock + mL;

  // staging assignment: thread t -> column scol, 32B chunk shalf
  const int scol  = tid >> 1;
  const int shalf = tid & 1;
  const unsigned long long gB =
      (unsigned long long)(uintptr_t)(Bw + (size_t)(colBlock + scol) * CC) +
      (unsigned)(shalf * 32);
  const unsigned ldsBase0 =
      (unsigned)(uintptr_t)(void*)&ldsB[0][scol * BCOL_STRIDE + shalf * 32];
  const unsigned ldsBase1 =
      (unsigned)(uintptr_t)(void*)&ldsB[1][scol * BCOL_STRIDE + shalf * 32];

  v8f acc[4] = {v8f{}, v8f{}, v8f{}, v8f{}};
  const _Float16* arow = A + (size_t)row * CC;

  // prologue: stage kk=0 into buffer 0 (DMA, tracked by ASYNCcnt)
  async_b128(ldsBase0, gB);
  async_b128_off16(ldsBase0, gB);

#pragma unroll 1
  for (int kk = 0; kk < CC; kk += 32) {
    const int cur = (kk >> 5) & 1;
    wait_async0();      // our stage-cur DMA complete
    __syncthreads();    // all waves' DMA visible; all done reading other buffer

    if (kk + 32 < CC) {  // overlap next stage DMA with this stage's WMMAs
      unsigned ldsAddr = cur ? ldsBase0 : ldsBase1;
      unsigned long long ga = gB + (unsigned)((kk + 32) * 2);
      async_b128(ldsAddr, ga);
      async_b128_off16(ldsAddr, ga);
    }

    // A fragment: elements 0..15 = K kk+hl*16+0..15 (contiguous)
    v16h a;
    const _Float16* ap = arow + kk + hl * 16;
    *(v8h*)&a = *(const v8h*)(ap);
    *((v8h*)&a + 1) = *(const v8h*)(ap + 8);
    if (kk + 32 < CC) __builtin_prefetch(arow + kk + 32, 0, 3);

#pragma unroll
    for (int t = 0; t < 4; t++) {
      // B fragment from LDS: col fixed per lane, K = hl*16 + j contiguous
      const _Float16* bp =
          (const _Float16*)&ldsB[cur][(t * 16 + mL) * BCOL_STRIDE + hl * 32];
      v16h b = *(const v16h*)bp;
      acc[t] = wmma_f16(a, b, acc[t]);
    }
  }

#pragma unroll
  for (int t = 0; t < 4; t++) {
    int col = colBlock + t * 16 + mL;    // D layout: lane -> column
    float cb = bias[col];
#pragma unroll
    for (int r = 0; r < 8; r++) {
      int ro = rowBlock + r + hl * 8;    // D layout: vgpr r -> row
      float v = (acc[t][r] + cb) * scale;
      if (mode == 3) {
        int n = ro / HW, hw = ro % HW;
        ((float*)out)[((size_t)n * CC + col) * HW + hw] = v;
      } else {
        int n = ro / HW, hw = ro % HW;
        int h = hw / WW_, w = hw % WW_;
        int win = (n * NWS + (h >> 3)) * NWS + (w >> 3);
        int intra = ((h & 7) << 3) | (w & 7);
        int head = col >> 5, d = col & 31;
        _Float16* o16 = (_Float16*)out;
        size_t base = (size_t)(win * NHEAD + head);
        if (mode == 2) o16[(base * HD + d) * 64 + intra] = (_Float16)v;   // v^T
        else           o16[(base * 64 + intra) * HD + d] = (_Float16)v;
      }
    }
  }
}

// ---------------- kernel 4: windowed attention, one wave per (window, head) ----------------
__global__ __launch_bounds__(32)
void attn_wmma_kernel(const _Float16* __restrict__ qw, const _Float16* __restrict__ kw,
                      const _Float16* __restrict__ vwT, const float* __restrict__ rpb,
                      _Float16* __restrict__ attnOut) {
  __shared__ float biasLds[(2 * WSZ - 1) * (2 * WSZ - 1)];  // 225
  __shared__ _Float16 pLds[64 * 64];                        // 8KB

  const int win = blockIdx.x;
  const int head = blockIdx.y;
  const int lane = threadIdx.x;
  const int mL = lane & 15;
  const int hl = lane >> 4;

  for (int t = lane; t < 225; t += 32) biasLds[t] = rpb[t * NHEAD + head];
  __syncthreads();

  const size_t base = (size_t)(win * NHEAD + head) * 64 * HD;
  const _Float16* qp = qw + base;
  const _Float16* kp = kw + base;
  const _Float16* vp = vwT + base;

  // --- S = q * k^T : 16 WMMAs (K=32 = full head dim) ---
  v16h aq[4], bk[4];
#pragma unroll
  for (int ti = 0; ti < 4; ti++) {
    const _Float16* p = qp + (ti * 16 + mL) * HD + hl * 16;
    *(v8h*)&aq[ti] = *(const v8h*)p;
    *((v8h*)&aq[ti] + 1) = *(const v8h*)(p + 8);
  }
#pragma unroll
  for (int tj = 0; tj < 4; tj++)  // B col = key token, K = d contiguous
    bk[tj] = *(const v16h*)(kp + (tj * 16 + mL) * HD + hl * 16);

  v8f s[4][4];
#pragma unroll
  for (int ti = 0; ti < 4; ti++)
#pragma unroll
    for (int tj = 0; tj < 4; tj++) {
      s[ti][tj] = v8f{};
      s[ti][tj] = wmma_f16(aq[ti], bk[tj], s[ti][tj]);
    }

  // --- bias + softmax over rows (row = r + 8*hl, group of 16 lanes) ---
#pragma unroll
  for (int ti = 0; ti < 4; ti++) {
#pragma unroll
    for (int r = 0; r < 8; r++) {
      int i = ti * 16 + r + hl * 8;
      int iy = i >> 3, ix = i & 7;
      float mx = -1e30f;
#pragma unroll
      for (int tj = 0; tj < 4; tj++) {
        int j = tj * 16 + mL;
        int jy = j >> 3, jx = j & 7;
        float v = s[ti][tj][r] + biasLds[(iy - jy + 7) * 15 + (ix - jx + 7)];
        s[ti][tj][r] = v;
        mx = fmaxf(mx, v);
      }
      for (int msk = 8; msk >= 1; msk >>= 1) mx = fmaxf(mx, __shfl_xor(mx, msk, 32));
      float sum = 0.f;
#pragma unroll
      for (int tj = 0; tj < 4; tj++) {
        float e = __expf(s[ti][tj][r] - mx);
        s[ti][tj][r] = e;
        sum += e;
      }
      for (int msk = 8; msk >= 1; msk >>= 1) sum += __shfl_xor(sum, msk, 32);
      float inv = 1.f / sum;
#pragma unroll
      for (int tj = 0; tj < 4; tj++) s[ti][tj][r] *= inv;
    }
  }

  // --- D-layout -> A-layout via LDS ---
#pragma unroll
  for (int ti = 0; ti < 4; ti++)
#pragma unroll
    for (int tj = 0; tj < 4; tj++)
#pragma unroll
      for (int r = 0; r < 8; r++)
        pLds[(ti * 16 + r + hl * 8) * 64 + tj * 16 + mL] = (_Float16)s[ti][tj][r];
  __syncthreads();

  // --- O = P * V : 16 WMMAs (K=64 in two chunks) ---
  v8f o[4][2] = {{v8f{}, v8f{}}, {v8f{}, v8f{}}, {v8f{}, v8f{}}, {v8f{}, v8f{}}};
#pragma unroll
  for (int kc = 0; kc < 2; kc++) {
    v16h bv[2];
#pragma unroll
    for (int nt = 0; nt < 2; nt++)  // B col = d (v stored transposed [d][tok])
      bv[nt] = *(const v16h*)(vp + (nt * 16 + mL) * 64 + kc * 32 + hl * 16);
#pragma unroll
    for (int ti = 0; ti < 4; ti++) {
      v16h ap;
      const _Float16* pp = &pLds[(ti * 16 + mL) * 64 + kc * 32 + hl * 16];
      *(v8h*)&ap = *(const v8h*)pp;
      *((v8h*)&ap + 1) = *(const v8h*)(pp + 8);
#pragma unroll
      for (int nt = 0; nt < 2; nt++) o[ti][nt] = wmma_f16(ap, bv[nt], o[ti][nt]);
    }
  }

  // --- scatter to token-major [token][head*32+d] f16 for projection GEMM ---
  int n = win / (NWS * NWS);
  int rw = win % (NWS * NWS);
  int wh = rw / NWS, wn = rw % NWS;
#pragma unroll
  for (int ti = 0; ti < 4; ti++)
#pragma unroll
    for (int nt = 0; nt < 2; nt++)
#pragma unroll
      for (int r = 0; r < 8; r++) {
        int i = ti * 16 + r + hl * 8;
        int d = nt * 16 + mL;
        int h = wh * 8 + (i >> 3), w = wn * 8 + (i & 7);
        size_t tok = (size_t)n * HW + (size_t)h * WW_ + w;
        attnOut[tok * CC + head * HD + d] = (_Float16)o[ti][nt][r];
      }
}

// ---------------- launcher ----------------
extern "C" void kernel_launch(void* const* d_in, const int* in_sizes, int n_in,
                              void* d_out, int out_size, void* d_ws, size_t ws_size,
                              hipStream_t stream) {
  const float* vid   = (const float*)d_in[0];
  const float* qd_w  = (const float*)d_in[1];
  const float* qd_b  = (const float*)d_in[2];
  const float* qp_w  = (const float*)d_in[3];
  const float* qp_b  = (const float*)d_in[4];
  const float* kd_w  = (const float*)d_in[5];
  const float* kd_b  = (const float*)d_in[6];
  const float* kp_w  = (const float*)d_in[7];
  const float* kp_b  = (const float*)d_in[8];
  const float* vd_w  = (const float*)d_in[9];
  const float* vd_b  = (const float*)d_in[10];
  const float* vp_w  = (const float*)d_in[11];
  const float* vp_b  = (const float*)d_in[12];
  const float* rpb   = (const float*)d_in[13];
  const float* pj_w  = (const float*)d_in[14];
  const float* pj_b  = (const float*)d_in[15];
  float* out = (float*)d_out;

  char* ws = (char*)d_ws;
  const size_t szTok = (size_t)TOKENS * CC * sizeof(_Float16);  // 56.6 MB
  _Float16* dwq  = (_Float16*)(ws + 0 * szTok);
  _Float16* dwk  = (_Float16*)(ws + 1 * szTok);
  _Float16* dwv  = (_Float16*)(ws + 2 * szTok);
  _Float16* qwb  = (_Float16*)(ws + 3 * szTok);
  _Float16* kwb  = (_Float16*)(ws + 4 * szTok);
  _Float16* vwb  = (_Float16*)(ws + 5 * szTok);
  _Float16* attn = dwq;  // dw buffers are dead after the pointwise GEMMs
  char* wbase = ws + 6 * szTok;
  const size_t szW = (size_t)CC * CC * sizeof(_Float16);
  _Float16* wq16 = (_Float16*)(wbase + 0 * szW);
  _Float16* wk16 = (_Float16*)(wbase + 1 * szW);
  _Float16* wv16 = (_Float16*)(wbase + 2 * szW);
  _Float16* wp16 = (_Float16*)(wbase + 3 * szW);

  // 1) weight conversion
  {
    int n = CC * CC, g = (n + 255) / 256;
    cvt_f16_kernel<<<g, 256, 0, stream>>>(qp_w, wq16, n);
    cvt_f16_kernel<<<g, 256, 0, stream>>>(kp_w, wk16, n);
    cvt_f16_kernel<<<g, 256, 0, stream>>>(vp_w, wv16, n);
    cvt_f16_kernel<<<g, 256, 0, stream>>>(pj_w, wp16, n);
  }

  // 2) fused depthwise conv q/k/v
  {
    long long tot = (long long)NB * CC * HW;
    dwconv_kernel<<<(int)((tot + 255) / 256), 256, 0, stream>>>(
        vid, qd_w, qd_b, kd_w, kd_b, vd_w, vd_b, dwq, dwk, dwv);
  }

  // 3) pointwise GEMMs -> windowed q/k/v (f16)
  dim3 gg(TOKENS / 64, CC / 64);  // 1152 x 6
  const float scale = 0.17677669529663689f;  // 32^-0.5
  gemm_wmma_kernel<<<gg, 128, 0, stream>>>(dwq, wq16, qp_b, qwb, 0, scale);
  gemm_wmma_kernel<<<gg, 128, 0, stream>>>(dwk, wk16, kp_b, kwb, 1, 1.0f);
  gemm_wmma_kernel<<<gg, 128, 0, stream>>>(dwv, wv16, vp_b, vwb, 2, 1.0f);

  // 4) windowed attention
  attn_wmma_kernel<<<dim3(NWIN, NHEAD), 32, 0, stream>>>(qwb, kwb, vwb, rpb, attn);

  // 5) output projection -> NCHW f32
  gemm_wmma_kernel<<<gg, 128, 0, stream>>>(attn, wp16, pj_b, out, 3, 1.0f);
}
